// Discriminator_36507222016249
// MI455X (gfx1250) — matrix operand order repaired and checked by reading, not compiled, not run on hardware
//
#include <hip/hip_runtime.h>
#include <hip/hip_bf16.h>

// ---------------------------------------------------------------------------
// Problem constants (from reference)
// ---------------------------------------------------------------------------
#define B_    64
#define NS_   16
#define NK_   32
#define DK_   512
#define DIM_FT_  2048
#define DIM_HID_ 1024
#define NKDK_ (NK_ * DK_)          // 16384

typedef __attribute__((ext_vector_type(16))) __bf16 v16bf;
typedef __attribute__((ext_vector_type(8)))  float  v8f;
typedef int v4i_ __attribute__((vector_size(16)));   // builtin's pointee type

// ---------------------------------------------------------------------------
// CDNA5 async global->LDS copy (ASYNCcnt path). Probe-confirmed: the builtin
// exists with 4 args and v4i* pointer params on this toolchain.
// ---------------------------------------------------------------------------
#if defined(__HIP_DEVICE_COMPILE__) && defined(__gfx1250__)
  #if __has_builtin(__builtin_amdgcn_global_load_async_to_lds_b128)
    #define HAVE_ASYNC_LDS 1
    #define ASYNC_CP_B128(gsrc, ldst) \
        __builtin_amdgcn_global_load_async_to_lds_b128((v4i_*)(gsrc), (v4i_*)(ldst), 0, 0)
  #elif __has_builtin(__builtin_amdgcn_global_async_load_to_lds_b128)
    #define HAVE_ASYNC_LDS 1
    #define ASYNC_CP_B128(gsrc, ldst) \
        __builtin_amdgcn_global_async_load_to_lds_b128((v4i_*)(gsrc), (v4i_*)(ldst), 0, 0)
  #else
    #define HAVE_ASYNC_LDS 0
  #endif
#else
  #define HAVE_ASYNC_LDS 0
#endif

// ---------------------------------------------------------------------------
// bf16 helpers (round-to-nearest-even), stored as unsigned short
// ---------------------------------------------------------------------------
__device__ __forceinline__ unsigned short f2bf(float f) {
    union { float f; unsigned u; } v; v.f = f;
    unsigned u = v.u;
    unsigned r = (u + 0x7FFFu + ((u >> 16) & 1u)) >> 16;
    return (unsigned short)r;
}
__device__ __forceinline__ float bf2f(unsigned short h) {
    union { unsigned u; float f; } v; v.u = ((unsigned)h) << 16;
    return v.f;
}

// ---------------------------------------------------------------------------
// Elementwise fp32 -> bf16 (4 elems / thread)
// ---------------------------------------------------------------------------
__global__ void k_f32_to_bf16(const float* __restrict__ src,
                              unsigned short* __restrict__ dst, int n) {
    int i = (blockIdx.x * 256 + threadIdx.x) * 4;
    if (i + 3 < n) {
        float4 v = *(const float4*)(src + i);
        uint2 r;
        r.x = (unsigned)f2bf(v.x) | ((unsigned)f2bf(v.y) << 16);
        r.y = (unsigned)f2bf(v.z) | ((unsigned)f2bf(v.w) << 16);
        *(uint2*)(dst + i) = r;
    }
}

// ---------------------------------------------------------------------------
// W (K x N, fp32, row-major) -> Wt (N x K, bf16, row-major); 64x64 LDS tiles
// ---------------------------------------------------------------------------
__global__ __launch_bounds__(256) void k_transpose_f32_to_bf16(
    const float* __restrict__ W, unsigned short* __restrict__ Wt, int K, int N) {
    __shared__ float tile[64][65];
    const int k0 = blockIdx.y * 64;
    const int n0 = blockIdx.x * 64;
    const int t  = threadIdx.x;
    for (int i = t; i < 64 * 64; i += 256) {
        int kk = i >> 6, nn = i & 63;
        tile[kk][nn] = W[(size_t)(k0 + kk) * N + (n0 + nn)];
    }
    __syncthreads();
    for (int i = t; i < 64 * 64; i += 256) {
        int nn = i >> 6, kk = i & 63;
        Wt[(size_t)(n0 + nn) * K + (k0 + kk)] = f2bf(tile[kk][nn]);
    }
}

// ---------------------------------------------------------------------------
// bf16 GEMM:  C(MxN, bf16) = A(MxK, bf16) @ Bt(NxK, bf16)^T
// Block tile 64x256, 8 waves, each wave 32x64 (2x4 WMMA 16x16 tiles), KT=32.
// LDS rows padded to 40 bf16 (80B) -> conflict-free ds_load_b128.
// Async path: double-buffered LDS fed by global_load_async_to_lds_b128,
// synchronized with s_wait_asynccnt / s_wait_dscnt + workgroup barriers.
// ---------------------------------------------------------------------------
#define BM 64
#define BN 256
#define KT 32
#define LDA 40

union Frag { uint4 q[2]; v16bf v; };

__global__ __launch_bounds__(256) void k_gemm_bf16(
    const unsigned short* __restrict__ A,
    const unsigned short* __restrict__ Bt,
    unsigned short* __restrict__ Cbf,
    int M, int N, int K) {
#if HAVE_ASYNC_LDS
    __shared__ unsigned short lds_a[2][BM * LDA];
    __shared__ unsigned short lds_b[2][BN * LDA];
#else
    __shared__ unsigned short lds_a[1][BM * LDA];
    __shared__ unsigned short lds_b[1][BN * LDA];
#endif

    const int t    = threadIdx.x;
    const int lane = t & 31;
    const int wave = t >> 5;
    const int hi   = lane >> 4;
    const int lq   = lane & 15;
    const int wm   = wave & 1;   // 2 M-groups of 32 rows
    const int wn   = wave >> 1;  // 4 N-groups of 64 cols
    const int m0   = blockIdx.y * BM;
    const int n0   = blockIdx.x * BN;

    v8f acc[2][4];
#pragma unroll
    for (int i = 0; i < 2; ++i)
#pragma unroll
        for (int j = 0; j < 4; ++j)
#pragma unroll
            for (int r = 0; r < 8; ++r) acc[i][j][r] = 0.0f;

    const int aRow = t >> 2;          // 0..63
    const int aCol = (t & 3) << 3;    // 0,8,16,24

    const unsigned short* gA = A  + (size_t)(m0 + aRow) * K + aCol;  // + k0
    const unsigned short* gB = Bt + (size_t)(n0 + t) * K;            // + k0

#if HAVE_ASYNC_LDS
    // ---- prologue: fill buffer 0 ----
    ASYNC_CP_B128(gA, &lds_a[0][aRow * LDA + aCol]);
#pragma unroll
    for (int c = 0; c < 4; ++c)
        ASYNC_CP_B128(gB + c * 8, &lds_b[0][t * LDA + c * 8]);

    int buf = 0;
    for (int k0 = 0; k0 < K; k0 += KT) {
        // all my ds reads of the (about to be overwritten) other buffer done
        asm volatile("s_wait_dscnt 0x0" ::: "memory");
        __syncthreads();

        if (k0 + KT < K) {  // issue copies for the next K-tile into other buf
            const int nb = buf ^ 1;
            ASYNC_CP_B128(gA + k0 + KT, &lds_a[nb][aRow * LDA + aCol]);
#pragma unroll
            for (int c = 0; c < 4; ++c)
                ASYNC_CP_B128(gB + k0 + KT + c * 8, &lds_b[nb][t * LDA + c * 8]);
            // current buffer's 5 copies (issued last iter) have retired
            asm volatile("s_wait_asynccnt 0x5" ::: "memory");
        } else {
            asm volatile("s_wait_asynccnt 0x0" ::: "memory");
        }
        __syncthreads();  // current tile visible to all waves

        const unsigned short* la = lds_a[buf];
        const unsigned short* lb = lds_b[buf];
#else
    for (int k0 = 0; k0 < K; k0 += KT) {
        // ---- global -> VGPR -> LDS (A: 64x32, B: 256x32) ----
        {
            uint4 v = *(const uint4*)(gA + k0);
            *(uint4*)(lds_a[0] + aRow * LDA + aCol) = v;
        }
#pragma unroll
        for (int c = 0; c < 4; ++c) {
            uint4 v = *(const uint4*)(gB + k0 + c * 8);
            *(uint4*)(lds_b[0] + t * LDA + c * 8) = v;
        }
        if (k0 + KT < K) {   // prefetch next K-tile -> global_prefetch_b8
            __builtin_prefetch(gA + k0 + KT, 0, 1);
            __builtin_prefetch(gB + k0 + KT, 0, 1);
        }
        __syncthreads();
        const unsigned short* la = lds_a[0];
        const unsigned short* lb = lds_b[0];
#endif

        // ---- LDS -> fragments ----
        Frag af[2], bfr[4];
#pragma unroll
        for (int i = 0; i < 2; ++i) {
            const unsigned short* p = la + (wm * 32 + i * 16 + lq) * LDA + hi * 8;
            af[i].q[0] = *(const uint4*)p;          // K = hi*8 .. +8
            af[i].q[1] = *(const uint4*)(p + 16);   // K = 16+hi*8 .. +8
        }
#pragma unroll
        for (int j = 0; j < 4; ++j) {
            const unsigned short* p = lb + (wn * 64 + j * 16 + lq) * LDA + hi * 8;
            bfr[j].q[0] = *(const uint4*)p;         // K = hi*8 .. +8
            bfr[j].q[1] = *(const uint4*)(p + 16);  // K = 16+hi*8 .. +8
        }
        // ---- 8 WMMAs ----
#pragma unroll
        for (int i = 0; i < 2; ++i)
#pragma unroll
            for (int j = 0; j < 4; ++j)
                acc[i][j] = __builtin_amdgcn_wmma_f32_16x16x32_bf16(
                    false, af[i].v, false, bfr[j].v, (short)0, acc[i][j], false, false);

#if HAVE_ASYNC_LDS
        buf ^= 1;
#else
        __syncthreads();
#endif
    }

    // ---- epilogue: store bf16 C ----
#pragma unroll
    for (int i = 0; i < 2; ++i)
#pragma unroll
        for (int j = 0; j < 4; ++j)
#pragma unroll
            for (int r = 0; r < 8; ++r) {
                int row = m0 + wm * 32 + i * 16 + 8 * hi + r;
                int col = n0 + wn * 64 + j * 16 + lq;
                Cbf[(size_t)row * N + col] = f2bf(acc[i][j][r]);
            }
}

// ---------------------------------------------------------------------------
// sent2sent: one wave per (b,k).  Gram = s2s_tile(16x512) @ s2s_tile^T via
// 16 accumulated WMMAs; tanh; column-sum minus diagonal; /15 -> dist2sent.
// ---------------------------------------------------------------------------
__global__ __launch_bounds__(256) void k_sent2sent(
    const unsigned short* __restrict__ s2s, float* __restrict__ d2s) {
    const int t    = threadIdx.x;
    const int lane = t & 31;
    const int wid  = blockIdx.x * 8 + (t >> 5);  // 0..2047
    const int b    = wid >> 5;                   // /NK
    const int k    = wid & 31;
    const int hi   = lane >> 4;
    const int lq   = lane & 15;

    v8f acc;
#pragma unroll
    for (int r = 0; r < 8; ++r) acc[r] = 0.0f;

    // row s = lq of the (16 x 512) tile for this (b,k)
    const unsigned short* base = s2s + (size_t)(b * NS_ + lq) * NKDK_ + k * DK_;
    for (int d0 = 0; d0 < DK_; d0 += KT) {
        Frag fa, fb;
        const unsigned short* pa = base + d0 + hi * 8;
        fa.q[0] = *(const uint4*)pa;
        fa.q[1] = *(const uint4*)(pa + 16);
        const unsigned short* pb = base + d0 + hi * 8;  // B-fragment layout
        fb.q[0] = *(const uint4*)pb;
        fb.q[1] = *(const uint4*)(pb + 16);
        acc = __builtin_amdgcn_wmma_f32_16x16x32_bf16(
            false, fa.v, false, fb.v, (short)0, acc, false, false);
    }

    const float inv = 0.04419417382415922f;  // 1/sqrt(512)
    float partial = 0.0f, diag = 0.0f;
#pragma unroll
    for (int r = 0; r < 8; ++r) {
        int Mrow = r + 8 * hi;
        float v = tanhf(acc[r] * inv);
        partial += v;
        if (Mrow == lq) diag = v;  // element (lq, lq)
    }
    partial += __shfl_xor(partial, 16, 32);
    diag    += __shfl_xor(diag, 16, 32);
    float res = (partial - diag) * (1.0f / 15.0f);
    if (hi == 0)  // lanes 0..15 write s = lq
        d2s[(size_t)b * (NS_ * NK_) + lq * NK_ + k] = res;
}

// ---------------------------------------------------------------------------
// dist2img: one wave per (b,s); lane = k; serial 512-dot per lane (bf16)
// ---------------------------------------------------------------------------
__global__ __launch_bounds__(256) void k_dist2img(
    const unsigned short* __restrict__ img_e,
    const unsigned short* __restrict__ s2i,
    float* __restrict__ d2i) {
    const int t    = threadIdx.x;
    const int lane = t & 31;                    // k
    const int wid  = blockIdx.x * 8 + (t >> 5); // 0..1023 = b*NS+s
    const int b    = wid >> 4;

    const unsigned short* pi = img_e + (size_t)b * NKDK_ + lane * DK_;
    const unsigned short* ps = s2i   + (size_t)wid * NKDK_ + lane * DK_;
    float sum = 0.0f;
    for (int d0 = 0; d0 < DK_; d0 += 8) {
        uint4 a = *(const uint4*)(pi + d0);
        uint4 c = *(const uint4*)(ps + d0);
        const unsigned* au = (const unsigned*)&a;
        const unsigned* cu = (const unsigned*)&c;
#pragma unroll
        for (int q = 0; q < 4; ++q) {
            sum = fmaf(bf2f((unsigned short)(au[q] & 0xffffu)),
                       bf2f((unsigned short)(cu[q] & 0xffffu)), sum);
            sum = fmaf(bf2f((unsigned short)(au[q] >> 16)),
                       bf2f((unsigned short)(cu[q] >> 16)), sum);
        }
    }
    const float inv = 0.04419417382415922f;
    d2i[(size_t)wid * NK_ + lane] = tanhf(sum * inv);
}

// ---------------------------------------------------------------------------
// Final 32->2 feed-forward heads + sum
// ---------------------------------------------------------------------------
__global__ void k_ff(const float* __restrict__ d2i, const float* __restrict__ d2s,
                     const float* __restrict__ wi, const float* __restrict__ bi,
                     const float* __restrict__ ws, const float* __restrict__ bs,
                     float* __restrict__ out, float* __restrict__ out_img,
                     float* __restrict__ out_sent) {
    int idx = blockIdx.x * 256 + threadIdx.x;  // 0..2047
    if (idx >= 2048) return;
    int row = idx >> 1, c = idx & 1;
    float si = bi[c], ss = bs[c];
#pragma unroll
    for (int k = 0; k < NK_; ++k) {
        si = fmaf(d2i[row * NK_ + k], wi[k * 2 + c], si);
        ss = fmaf(d2s[row * NK_ + k], ws[k * 2 + c], ss);
    }
    out_img[idx]  = si;
    out_sent[idx] = ss;
    out[idx]      = si + ss;
}

// ---------------------------------------------------------------------------
// Host-side orchestration
// ---------------------------------------------------------------------------
extern "C" void kernel_launch(void* const* d_in, const int* in_sizes, int n_in,
                              void* d_out, int out_size, void* d_ws, size_t ws_size,
                              hipStream_t stream) {
    const float* fts      = (const float*)d_in[0];
    const float* hidden   = (const float*)d_in[1];
    const float* W_img    = (const float*)d_in[2];
    const float* W_si     = (const float*)d_in[3];
    const float* W_ss     = (const float*)d_in[4];
    const float* ff_img_w = (const float*)d_in[5];
    const float* ff_img_b = (const float*)d_in[6];
    const float* ff_sent_w= (const float*)d_in[7];
    const float* ff_sent_b= (const float*)d_in[8];
    float* out = (float*)d_out;

    // workspace layout (bytes)
    char* ws = (char*)d_ws;
    unsigned short* wt   = (unsigned short*)(ws);              // 64 MB (reused per GEMM)
    unsigned short* ftsb = (unsigned short*)(ws + 67108864);   // 256 KB
    unsigned short* hidb = (unsigned short*)(ws + 67371008);   // 2 MB
    unsigned short* imgb = (unsigned short*)(ws + 69468160);   // 2 MB  (img_e bf16)
    unsigned short* s2ib = (unsigned short*)(ws + 71565312);   // 32 MB (s2i bf16)
    unsigned short* s2sb = (unsigned short*)(ws + 105119744);  // 32 MB (s2s bf16)

    float* out_img  = out + 2048;
    float* out_sent = out + 4096;
    float* d2i      = out + 6144;
    float* d2s      = out + 38912;

    // input conversions
    k_f32_to_bf16<<<(B_ * DIM_FT_) / 1024, 256, 0, stream>>>(fts, ftsb, B_ * DIM_FT_);
    k_f32_to_bf16<<<(B_ * NS_ * DIM_HID_) / 1024, 256, 0, stream>>>(hidden, hidb, B_ * NS_ * DIM_HID_);

    // GEMM 1: img_e = fts @ W_img   (64 x 16384, K=2048)
    k_transpose_f32_to_bf16<<<dim3(NKDK_ / 64, DIM_FT_ / 64), 256, 0, stream>>>(W_img, wt, DIM_FT_, NKDK_);
    k_gemm_bf16<<<dim3(NKDK_ / BN, B_ / BM), 256, 0, stream>>>(ftsb, wt, imgb, B_, NKDK_, DIM_FT_);

    // GEMM 2: s2i = hidden @ W_si   (1024 x 16384, K=1024)
    k_transpose_f32_to_bf16<<<dim3(NKDK_ / 64, DIM_HID_ / 64), 256, 0, stream>>>(W_si, wt, DIM_HID_, NKDK_);
    k_gemm_bf16<<<dim3(NKDK_ / BN, (B_ * NS_) / BM), 256, 0, stream>>>(hidb, wt, s2ib, B_ * NS_, NKDK_, DIM_HID_);

    // GEMM 3: s2s = hidden @ W_ss
    k_transpose_f32_to_bf16<<<dim3(NKDK_ / 64, DIM_HID_ / 64), 256, 0, stream>>>(W_ss, wt, DIM_HID_, NKDK_);
    k_gemm_bf16<<<dim3(NKDK_ / BN, (B_ * NS_) / BM), 256, 0, stream>>>(hidb, wt, s2sb, B_ * NS_, NKDK_, DIM_HID_);

    // reductions
    k_sent2sent<<<(B_ * NK_) / 8, 256, 0, stream>>>(s2sb, d2s);
    k_dist2img<<<(B_ * NS_) / 8, 256, 0, stream>>>(imgb, s2ib, d2i);
    k_ff<<<8, 256, 0, stream>>>(d2i, d2s, ff_img_w, ff_img_b, ff_sent_w, ff_sent_b,
                                out, out_img, out_sent);
}
